// MABNet_52295521796557
// MI455X (gfx1250) — matrix operand order repaired and checked by reading, not compiled, not run on hardware
//
#include <hip/hip_runtime.h>

typedef _Float16 f16;
typedef __attribute__((ext_vector_type(16))) _Float16 v16h;
typedef __attribute__((ext_vector_type(8)))  _Float16 v8h;
typedef __attribute__((ext_vector_type(8)))  float    v8f;

#define DEV __device__ __forceinline__

// ---------------- problem constants ----------------
// B=8, N=256, BN=2048, E=16384, H=256, NH=8, HD=32, VD=8, CUTOFF=5
#define C_BN   2048
#define C_E    16384
#define C_H    256
#define C_PI_OVER_CUT 0.6283185307179586f

DEV float silu_f(float x) { return x / (1.0f + __expf(-x)); }

// ---------------- WMMA tile helpers (16x16 tile, K-loop of 32) ----------------
// A: row-major, lane (l&15) owns row; per-lane two contiguous 16B chunks per k-step.
// Bt: (N x K) row-major (i.e. B transposed); lane (l&15) owns column; two 16B chunks per k-step.
DEV v8f wmma_acc(const f16* __restrict__ arow, const f16* __restrict__ Bt, int ldb,
                 int col0, int K, int lane, v8f acc) {
  const f16* bcol = Bt + (size_t)(col0 + (lane & 15)) * ldb + ((lane & 16) ? 8 : 0);
  const int koffA = (lane & 16) ? 8 : 0;
  #pragma unroll 4
  for (int kb = 0; kb < K; kb += 32) {
    v8h lo = *(const v8h*)(arow + kb + koffA);
    v8h hi = *(const v8h*)(arow + kb + 16 + koffA);
    v16h a = __builtin_shufflevector(lo, hi, 0,1,2,3,4,5,6,7,8,9,10,11,12,13,14,15);
    v8h blo = *(const v8h*)(bcol + kb);
    v8h bhi = *(const v8h*)(bcol + kb + 16);
    v16h b = __builtin_shufflevector(blo, bhi, 0,1,2,3,4,5,6,7,8,9,10,11,12,13,14,15);
    acc = __builtin_amdgcn_wmma_f32_16x16x32_f16(false, a, false, b, (short)0, acc,
                                                 false, false);
  }
  return acc;
}

DEV v8f wmma_tile(const f16* __restrict__ A, int lda, int row0,
                  const f16* __restrict__ Bt, int ldb, int col0, int K, int lane) {
  v8f acc = {0.f,0.f,0.f,0.f,0.f,0.f,0.f,0.f};
  const f16* arow = A + (size_t)(row0 + (lane & 15)) * lda;
  return wmma_acc(arow, Bt, ldb, col0, K, lane, acc);
}

// ---------------- utility kernels ----------------
__global__ void k_zero(float* p, int n) {
  int i = blockIdx.x * 256 + threadIdx.x;
  if (i < n) p[i] = 0.0f;
}

__global__ void k_cvt(const float* __restrict__ s, f16* __restrict__ d, int n) {
  int i = blockIdx.x * 256 + threadIdx.x;
  if (i < n) d[i] = (f16)s[i];
}

// W (K x Nw) f32 -> Wt (Nw x K) f16
__global__ void k_cvtT(const float* __restrict__ W, f16* __restrict__ Wt, int K, int Nw) {
  int idx = blockIdx.x * 256 + threadIdx.x;
  int k = idx / Nw, n = idx - k * Nw;
  Wt[(size_t)n * K + k] = (f16)W[idx];
}

// LayerNorm: one block (256 threads) per node row
__global__ void k_ln(const float* __restrict__ x, const float* __restrict__ w,
                     const float* __restrict__ b, f16* __restrict__ out) {
  __shared__ float sbuf[256];
  int n = blockIdx.x, h = threadIdx.x;
  float v = x[(size_t)n * C_H + h];
  sbuf[h] = v; __syncthreads();
  for (int s = 128; s > 0; s >>= 1) { if (h < s) sbuf[h] += sbuf[h + s]; __syncthreads(); }
  float mu = sbuf[0] * (1.0f / C_H); __syncthreads();
  float dv = v - mu;
  sbuf[h] = dv * dv; __syncthreads();
  for (int s = 128; s > 0; s >>= 1) { if (h < s) sbuf[h] += sbuf[h + s]; __syncthreads(); }
  float var = sbuf[0] * (1.0f / C_H);
  out[(size_t)n * C_H + h] = (f16)(dv * rsqrtf(var + 1e-5f) * w[h] + b[h]);
}

// ---------------- GEMM kernels (8 waves/block, 1 tile/wave) ----------------
// plain: out f16 (M x 256), K = 256
__global__ void k_gemm_f16(const f16* __restrict__ A, const f16* __restrict__ Bt,
                           f16* __restrict__ out, int M, int Nc) {
  int wid = (blockIdx.x * 256 + threadIdx.x) >> 5;
  int lane = threadIdx.x & 31;
  int ntn = Nc >> 4;
  int mt = wid / ntn, nt = wid - mt * ntn;
  v8f acc = wmma_tile(A, 256, mt * 16, Bt, 256, nt * 16, 256, lane);
  int col = nt * 16 + (lane & 15);
  int rbase = mt * 16 + ((lane >> 4) << 3);
  #pragma unroll
  for (int r = 0; r < 8; r++)
    out[(size_t)(rbase + r) * Nc + col] = (f16)acc[r];
}

// xn @ WvT, scattered into vvT[b][h][d][j]  (for attn@v B-operand)
__global__ void k_gemm_vvT(const f16* __restrict__ A, const f16* __restrict__ Bt,
                           f16* __restrict__ vvT) {
  int wid = (blockIdx.x * 256 + threadIdx.x) >> 5;
  int lane = threadIdx.x & 31;
  int mt = wid >> 4, nt = wid & 15;
  v8f acc = wmma_tile(A, 256, mt * 16, Bt, 256, nt * 16, 256, lane);
  int col = nt * 16 + (lane & 15);       // feature c = h*32+d
  int hh = col >> 5, d = col & 31;
  int rbase = mt * 16 + ((lane >> 4) << 3);
  #pragma unroll
  for (int r = 0; r < 8; r++) {
    int g = rbase + r;                   // node = b*256 + j
    int bb = g >> 8, j = g & 255;
    vvT[(((size_t)(bb * 8 + hh)) * 32 + d) * 256 + j] = (f16)acc[r];
  }
}

// attn[b,h,i,j] = silu(q.k / sqrt(32)), stored f16
__global__ void k_attn(const f16* __restrict__ q, const f16* __restrict__ k,
                       f16* __restrict__ attn) {
  int wid = (blockIdx.x * 256 + threadIdx.x) >> 5;
  int lane = threadIdx.x & 31;
  int bh = wid >> 8, rest = wid & 255;
  int it = rest >> 4, jt = rest & 15;
  int bb = bh >> 3, hh = bh & 7;
  const f16* Abase = q + (size_t)bb * 65536 + hh * 32;
  const f16* Bbase = k + (size_t)bb * 65536 + hh * 32;
  v8f acc = wmma_tile(Abase, 256, it * 16, Bbase, 256, jt * 16, 32, lane);
  int colj = jt * 16 + (lane & 15);
  int rbase = it * 16 + ((lane >> 4) << 3);
  #pragma unroll
  for (int r = 0; r < 8; r++) {
    float v = silu_f(acc[r] * 0.17677669529663687f);
    attn[((size_t)bh * 256 + (rbase + r)) * 256 + colj] = (f16)v;
  }
}

// v[b,i,h,d] = (attn @ vv)/N  -> v_h (BN x H) f16
__global__ void k_av(const f16* __restrict__ attn, const f16* __restrict__ vvT,
                     f16* __restrict__ v_h) {
  int wid = (blockIdx.x * 256 + threadIdx.x) >> 5;
  int lane = threadIdx.x & 31;
  int bh = wid >> 5, rest = wid & 31;
  int it = rest >> 1, nt = rest & 1;
  int bb = bh >> 3, hh = bh & 7;
  v8f acc = wmma_tile(attn + (size_t)bh * 65536, 256, it * 16,
                      vvT + (size_t)bh * 8192, 256, nt * 16, 256, lane);
  int d = nt * 16 + (lane & 15);
  int rbase = it * 16 + ((lane >> 4) << 3);
  #pragma unroll
  for (int r = 0; r < 8; r++) {
    int i = rbase + r;
    v_h[((size_t)bb * 256 + i) * 256 + hh * 32 + d] = (f16)(acc[r] * (1.0f / 256.0f));
  }
}

// generic: out f32 = act(A @ Bt^T + bias), K=256, act: 0=none 1=silu
__global__ void k_gemm_ba(const f16* __restrict__ A, const f16* __restrict__ Bt,
                          const float* __restrict__ bias, float* __restrict__ out,
                          int M, int Nc, int act) {
  int wid = (blockIdx.x * 256 + threadIdx.x) >> 5;
  int lane = threadIdx.x & 31;
  int ntn = Nc >> 4;
  int mt = wid / ntn, nt = wid - mt * ntn;
  v8f acc = wmma_tile(A, 256, mt * 16, Bt, 256, nt * 16, 256, lane);
  int col = nt * 16 + (lane & 15);
  float bv = bias ? bias[col] : 0.0f;
  int rbase = mt * 16 + ((lane >> 4) << 3);
  #pragma unroll
  for (int r = 0; r < 8; r++) {
    float v = acc[r] + bv;
    if (act) v = silu_f(v);
    out[(size_t)(rbase + r) * Nc + col] = v;
  }
}

// dv = silu(f_ij@Wdv + bdv); v_j = v2[src]*cut*dv ; write v_j f16 + atomic x_agg
__global__ void k_vj(const f16* __restrict__ fij, const f16* __restrict__ WdvT,
                     const float* __restrict__ bdv, const float* __restrict__ v2,
                     const int* __restrict__ srcI, const int* __restrict__ dstI,
                     const float* __restrict__ r_ij, f16* __restrict__ vj_h,
                     float* __restrict__ xagg) {
  int wid = (blockIdx.x * 256 + threadIdx.x) >> 5;
  int lane = threadIdx.x & 31;
  int mt = wid >> 4, nt = wid & 15;
  v8f acc = wmma_tile(fij, 256, mt * 16, WdvT, 256, nt * 16, 256, lane);
  int col = nt * 16 + (lane & 15);
  float bv = bdv[col];
  int rbase = mt * 16 + ((lane >> 4) << 3);
  #pragma unroll
  for (int r = 0; r < 8; r++) {
    int e = rbase + r;
    float t = silu_f(acc[r] + bv);
    float rr = r_ij[e];
    float cut = (rr < 5.0f) ? 0.5f * (__cosf(C_PI_OVER_CUT * rr) + 1.0f) : 0.0f;
    float vj = v2[(size_t)srcI[e] * C_H + col] * cut * t;
    vj_h[(size_t)e * C_H + col] = (f16)vj;
    atomicAdd(&xagg[(size_t)dstI[e] * C_H + col], vj);
  }
}

// fused w_dot: one tile = 2 edges x 8 vd rows x 16 feature cols.
// w1 = rej(vec[dst]@Wtrg, d), w2 = rej(vec[src]@Wsrc, -d); wdot = sum_vd w1*w2
__global__ void k_wdot(const f16* __restrict__ vec_h, const f16* __restrict__ WtrgT,
                       const f16* __restrict__ WsrcT, const int* __restrict__ srcI,
                       const int* __restrict__ dstI, const float* __restrict__ d_ij,
                       float* __restrict__ wdot) {
  int wid = (blockIdx.x * 256 + threadIdx.x) >> 5;
  int lane = threadIdx.x & 31;
  int te = wid >> 4, ht = wid & 15;
  int e0 = te * 2, e1 = te * 2 + 1;
  int m = lane & 15;
  int ea = (m < 8) ? e0 : e1;
  int vd = m & 7;
  const f16* arow1 = vec_h + ((size_t)dstI[ea] * 8 + vd) * 256;
  const f16* arow2 = vec_h + ((size_t)srcI[ea] * 8 + vd) * 256;
  v8f z = {0.f,0.f,0.f,0.f,0.f,0.f,0.f,0.f};
  v8f a1 = wmma_acc(arow1, WtrgT, 256, ht * 16, 256, lane, z);
  v8f a2 = wmma_acc(arow2, WsrcT, 256, ht * 16, 256, lane, z);
  int eg = (lane & 16) ? e1 : e0;
  const float* dp = d_ij + (size_t)eg * 8;
  float d8[8];
  #pragma unroll
  for (int r = 0; r < 8; r++) d8[r] = dp[r];
  float p1 = 0.f, p2 = 0.f;
  #pragma unroll
  for (int r = 0; r < 8; r++) { p1 += a1[r] * d8[r]; p2 += a2[r] * d8[r]; }
  float wd = 0.f;
  #pragma unroll
  for (int r = 0; r < 8; r++)
    wd += (a1[r] - p1 * d8[r]) * (a2[r] - p2 * d8[r]);
  wdot[(size_t)eg * C_H + ht * 16 + (lane & 15)] = wd;
}

// df_ij = silu(f_ij@Wf + bf) * wdot
__global__ void k_df(const f16* __restrict__ fij, const f16* __restrict__ WfT,
                     const float* __restrict__ bf, const float* __restrict__ wdot,
                     float* __restrict__ dfout) {
  int wid = (blockIdx.x * 256 + threadIdx.x) >> 5;
  int lane = threadIdx.x & 31;
  int mt = wid >> 4, nt = wid & 15;
  v8f acc = wmma_tile(fij, 256, mt * 16, WfT, 256, nt * 16, 256, lane);
  int col = nt * 16 + (lane & 15);
  float bv = bf[col];
  int rbase = mt * 16 + ((lane >> 4) << 3);
  #pragma unroll
  for (int r = 0; r < 8; r++) {
    int e = rbase + r;
    dfout[(size_t)e * C_H + col] = silu_f(acc[r] + bv) * wdot[(size_t)e * C_H + col];
  }
}

// vec_j scatter: dvec_agg[dst] += vec[src]*s1 + s2*d_ij   (E*H threads, VD loop)
__global__ void k_scatter(const float* __restrict__ s12, const float* __restrict__ vec,
                          const float* __restrict__ d_ij, const int* __restrict__ srcI,
                          const int* __restrict__ dstI, float* __restrict__ dvec) {
  int idx = blockIdx.x * 256 + threadIdx.x;
  int e = idx >> 8, h = idx & 255;
  float s1 = s12[(size_t)e * 512 + h];
  float s2 = s12[(size_t)e * 512 + 256 + h];
  const float* vsrc = vec + (size_t)srcI[e] * 2048;
  float* dacc = dvec + (size_t)dstI[e] * 2048;
  #pragma unroll
  for (int vd = 0; vd < 8; vd++) {
    float val = vsrc[vd * 256 + h] * s1 + s2 * d_ij[(size_t)e * 8 + vd];
    atomicAdd(&dacc[vd * 256 + h], val);
  }
}

__global__ void k_vecsum(const float* __restrict__ vp, float* __restrict__ vsum) {
  int idx = blockIdx.x * 256 + threadIdx.x;       // BN*H
  int n = idx >> 8, h = idx & 255;
  float s = 0.f;
  #pragma unroll
  for (int vd = 0; vd < 8; vd++) s += vp[((size_t)n * 8 + vd) * 512 + h];
  vsum[idx] = s;
}

__global__ void k_final_dx(const float* __restrict__ o, const float* __restrict__ vsum,
                           float* __restrict__ dx) {
  int idx = blockIdx.x * 256 + threadIdx.x;       // BN*H
  int n = idx >> 8, h = idx & 255;
  dx[idx] = vsum[idx] * o[(size_t)n * 768 + 256 + h] + o[(size_t)n * 768 + 512 + h];
}

__global__ void k_final_dvec(const float* __restrict__ o, const float* __restrict__ vp,
                             float* __restrict__ dvec) {
  int idx = blockIdx.x * 256 + threadIdx.x;       // BN*VD*H
  int n = idx >> 11, rem = idx & 2047;
  int vd = rem >> 8, h = rem & 255;
  float vec3 = vp[((size_t)n * 8 + vd) * 512 + 256 + h];
  dvec[idx] += vec3 * o[(size_t)n * 768 + h];
}

// ---------------- workspace layout (bytes) ----------------
static constexpr size_t OFF_XN    = 0;                       // BN*H f16     1 MB
static constexpr size_t OFF_Q     = OFF_XN    + 1048576;     // 1 MB
static constexpr size_t OFF_K     = OFF_Q     + 1048576;     // 1 MB
static constexpr size_t OFF_VVT   = OFF_K     + 1048576;     // 1 MB
static constexpr size_t OFF_ATTN  = OFF_VVT   + 1048576;     // 8 MB
static constexpr size_t OFF_VH    = OFF_ATTN  + 8388608;     // 1 MB
static constexpr size_t OFF_V2    = OFF_VH    + 1048576;     // f32 2 MB
static constexpr size_t OFF_FIJ   = OFF_V2    + 2097152;     // f16 8 MB
static constexpr size_t OFF_VJ    = OFF_FIJ   + 8388608;     // f16 8 MB
static constexpr size_t OFF_XAGG  = OFF_VJ    + 8388608;     // f32 2 MB
static constexpr size_t OFF_XAGH  = OFF_XAGG  + 2097152;     // f16 1 MB
static constexpr size_t OFF_S12   = OFF_XAGH  + 1048576;     // f32 33.5 MB
static constexpr size_t OFF_VECH  = OFF_S12   + 33554432;    // f16 8 MB
static constexpr size_t OFF_VP    = OFF_VECH  + 8388608;     // f32 33.5 MB
static constexpr size_t OFF_VSUM  = OFF_VP    + 33554432;    // f32 2 MB
static constexpr size_t OFF_O     = OFF_VSUM  + 2097152;     // f32 6 MB
static constexpr size_t OFF_WDOT  = OFF_O     + 6291456;     // f32 16 MB
static constexpr size_t OFF_WQT   = OFF_WDOT  + 16777216;
static constexpr size_t OFF_WKT   = OFF_WQT   + 131072;
static constexpr size_t OFF_WVT   = OFF_WKT   + 131072;
static constexpr size_t OFF_WAOT  = OFF_WVT   + 131072;
static constexpr size_t OFF_WVECT = OFF_WAOT  + 131072;      // 256x512
static constexpr size_t OFF_WDVT  = OFF_WVECT + 262144;
static constexpr size_t OFF_WST   = OFF_WDVT  + 131072;      // 256x512
static constexpr size_t OFF_WOT   = OFF_WST   + 262144;      // 256x768
static constexpr size_t OFF_WFT   = OFF_WOT   + 393216;
static constexpr size_t OFF_WSRCT = OFF_WFT   + 131072;
static constexpr size_t OFF_WTRGT = OFF_WSRCT + 131072;

extern "C" void kernel_launch(void* const* d_in, const int* in_sizes, int n_in,
                              void* d_out, int out_size, void* d_ws, size_t ws_size,
                              hipStream_t stream) {
  (void)in_sizes; (void)n_in; (void)out_size; (void)ws_size;
  char* ws = (char*)d_ws;

  const float* x     = (const float*)d_in[0];
  const float* vec   = (const float*)d_in[1];
  const int*   srcI  = (const int*)d_in[2];
  const int*   dstI  = srcI + C_E;
  const float* r_ij  = (const float*)d_in[3];
  const float* f_ij  = (const float*)d_in[4];
  const float* d_ij  = (const float*)d_in[5];
  const float* ln_w  = (const float*)d_in[7];
  const float* ln_b  = (const float*)d_in[8];
  const float* Wq    = (const float*)d_in[9];
  const float* Wk    = (const float*)d_in[10];
  const float* Wv    = (const float*)d_in[11];
  const float* Wao   = (const float*)d_in[12];
  const float* Wvec  = (const float*)d_in[13];
  const float* Wdv   = (const float*)d_in[14];
  const float* bdv   = (const float*)d_in[15];
  const float* Ws    = (const float*)d_in[16];
  const float* bs    = (const float*)d_in[17];
  const float* Wo    = (const float*)d_in[18];
  const float* bo    = (const float*)d_in[19];
  const float* Wf    = (const float*)d_in[20];
  const float* bf    = (const float*)d_in[21];
  const float* Wsrc  = (const float*)d_in[22];
  const float* Wtrg  = (const float*)d_in[23];

  f16*   xn_h   = (f16*)(ws + OFF_XN);
  f16*   q_h    = (f16*)(ws + OFF_Q);
  f16*   k_h    = (f16*)(ws + OFF_K);
  f16*   vvT    = (f16*)(ws + OFF_VVT);
  f16*   attn_h = (f16*)(ws + OFF_ATTN);
  f16*   v_h    = (f16*)(ws + OFF_VH);
  float* v2     = (float*)(ws + OFF_V2);
  f16*   fij_h  = (f16*)(ws + OFF_FIJ);
  f16*   vj_h   = (f16*)(ws + OFF_VJ);
  float* xagg   = (float*)(ws + OFF_XAGG);
  f16*   xagg_h = (f16*)(ws + OFF_XAGH);
  float* s12    = (float*)(ws + OFF_S12);
  f16*   vec_h  = (f16*)(ws + OFF_VECH);
  float* vp     = (float*)(ws + OFF_VP);
  float* vsum   = (float*)(ws + OFF_VSUM);
  float* o_f    = (float*)(ws + OFF_O);
  float* wdot   = (float*)(ws + OFF_WDOT);
  f16*   WqT    = (f16*)(ws + OFF_WQT);
  f16*   WkT    = (f16*)(ws + OFF_WKT);
  f16*   WvT    = (f16*)(ws + OFF_WVT);
  f16*   WaoT   = (f16*)(ws + OFF_WAOT);
  f16*   WvecT  = (f16*)(ws + OFF_WVECT);
  f16*   WdvT   = (f16*)(ws + OFF_WDVT);
  f16*   WsT    = (f16*)(ws + OFF_WST);
  f16*   WoT    = (f16*)(ws + OFF_WOT);
  f16*   WfT    = (f16*)(ws + OFF_WFT);
  f16*   WsrcT  = (f16*)(ws + OFF_WSRCT);
  f16*   WtrgT  = (f16*)(ws + OFF_WTRGT);

  float* dx_out   = (float*)d_out;                 // BN*H
  float* dvec_out = dx_out + 524288;               // BN*VD*H
  float* df_out   = dvec_out + 4194304;            // E*H

  // --- init accumulators ---
  k_zero<<<2048, 256, 0, stream>>>(xagg, 524288);
  k_zero<<<16384, 256, 0, stream>>>(dvec_out, 4194304);

  // --- stage conversions ---
  k_ln<<<2048, 256, 0, stream>>>(x, ln_w, ln_b, xn_h);
  k_cvtT<<<256, 256, 0, stream>>>(Wq,   WqT,   256, 256);
  k_cvtT<<<256, 256, 0, stream>>>(Wk,   WkT,   256, 256);
  k_cvtT<<<256, 256, 0, stream>>>(Wv,   WvT,   256, 256);
  k_cvtT<<<256, 256, 0, stream>>>(Wao,  WaoT,  256, 256);
  k_cvtT<<<512, 256, 0, stream>>>(Wvec, WvecT, 256, 512);
  k_cvtT<<<256, 256, 0, stream>>>(Wdv,  WdvT,  256, 256);
  k_cvtT<<<512, 256, 0, stream>>>(Ws,   WsT,   256, 512);
  k_cvtT<<<768, 256, 0, stream>>>(Wo,   WoT,   256, 768);
  k_cvtT<<<256, 256, 0, stream>>>(Wf,   WfT,   256, 256);
  k_cvtT<<<256, 256, 0, stream>>>(Wsrc, WsrcT, 256, 256);
  k_cvtT<<<256, 256, 0, stream>>>(Wtrg, WtrgT, 256, 256);
  k_cvt<<<16384, 256, 0, stream>>>(f_ij, fij_h, 4194304);
  k_cvt<<<16384, 256, 0, stream>>>(vec,  vec_h, 4194304);

  // --- attention path ---
  k_gemm_f16<<<256, 256, 0, stream>>>(xn_h, WqT, q_h, C_BN, 256);
  k_gemm_f16<<<256, 256, 0, stream>>>(xn_h, WkT, k_h, C_BN, 256);
  k_gemm_vvT<<<256, 256, 0, stream>>>(xn_h, WvT, vvT);
  k_attn<<<2048, 256, 0, stream>>>(q_h, k_h, attn_h);
  k_av<<<256, 256, 0, stream>>>(attn_h, vvT, v_h);
  k_gemm_ba<<<256, 256, 0, stream>>>(v_h, WaoT, nullptr, v2, C_BN, 256, 0);

  // --- edge path: dv, v_j (+x_agg atomics), s12, vec scatter ---
  k_vj<<<2048, 256, 0, stream>>>(fij_h, WdvT, bdv, v2, srcI, dstI, r_ij, vj_h, xagg);
  k_gemm_ba<<<4096, 256, 0, stream>>>(vj_h, WsT, bs, s12, C_E, 512, 1);
  k_scatter<<<16384, 256, 0, stream>>>(s12, vec, d_ij, srcI, dstI, dvec_out);

  // --- node outputs ---
  k_cvt<<<2048, 256, 0, stream>>>(xagg, xagg_h, 524288);
  k_gemm_ba<<<768, 256, 0, stream>>>(xagg_h, WoT, bo, o_f, C_BN, 768, 0);
  k_gemm_ba<<<4096, 256, 0, stream>>>(vec_h, WvecT, nullptr, vp, C_BN * 8, 512, 0);
  k_vecsum<<<2048, 256, 0, stream>>>(vp, vsum);
  k_final_dx<<<2048, 256, 0, stream>>>(o_f, vsum, dx_out);
  k_final_dvec<<<16384, 256, 0, stream>>>(o_f, vp, dvec_out);

  // --- w_dot + df_ij (dominant GEMMs, fused gather + rejection) ---
  k_wdot<<<16384, 256, 0, stream>>>(vec_h, WtrgT, WsrcT, srcI, dstI, d_ij, wdot);
  k_df<<<2048, 256, 0, stream>>>(fij_h, WfT, bf, wdot, df_out);
}
